// SSA_867583394310
// MI455X (gfx1250) — compile-verified, matrix operand order checked by reading, not run on hardware
//
#include <hip/hip_runtime.h>
#include <hip/hip_bf16.h>


typedef _Float16 h16;
typedef __attribute__((ext_vector_type(16))) _Float16 v16h;
typedef __attribute__((ext_vector_type(8)))  _Float16 v8h;
typedef __attribute__((ext_vector_type(8)))  float    v8f;
typedef __attribute__((ext_vector_type(8)))  int      v8i;

constexpr int B_ = 32, C_ = 512, N_ = 1024;
constexpr float TH_BN   = 1.1f;
constexpr float TH_ATTN = 4.4f;

__device__ __forceinline__ void async_cp16(unsigned lds_off, const void* gbase, unsigned goff)
{
    asm volatile("global_load_async_to_lds_b128 %0, %1, %2"
                 :: "v"(lds_off), "v"(goff), "s"(gbase)
                 : "memory");
}
__device__ __forceinline__ void wait_async0()
{
    asm volatile("s_wait_asynccnt 0" ::: "memory");
}
__device__ __forceinline__ unsigned lds_addr(const void* p)
{
    return (unsigned)(unsigned long long)p;
}

template<bool ABATCH, int EPI, bool TRANSOUT>
__global__ __launch_bounds__(256) void gemm_wmma(const h16* __restrict__ A,
                                                 const h16* __restrict__ X,
                                                 h16* __restrict__ Y,
                                                 int M, int K, int Ncols)
{
    __shared__ h16 sA[2][128 * 40];
    __shared__ h16 sX[2][128 * 40];
    const int b    = blockIdx.z;
    const int m0   = blockIdx.y * 128;
    const int n0   = blockIdx.x * 128;
    const int t    = threadIdx.x;
    const int lane = t & 31;
    const int wave = t >> 5;
    const int wm   = wave >> 2;
    const int wn   = wave & 3;
    const int rrow = lane & 15;
    const int rsel = (lane < 16) ? 0 : 8;
    const size_t aBase = ABATCH ? (size_t)b * (size_t)M * (size_t)K : 0;

    const unsigned r0 = (unsigned)t >> 2,          s0 = ((unsigned)t & 3) * 16;
    const unsigned r1 = (unsigned)(t + 256) >> 2,  s1 = ((unsigned)(t + 256) & 3) * 16;
    const unsigned kb = (unsigned)K * 2;

    auto issue = [&](int buf, int k0) {
        const char* gA = (const char*)(A + aBase + (size_t)m0 * K + k0);
        const char* gX = (const char*)(X + ((size_t)b * Ncols + n0) * K + k0);
        unsigned la = lds_addr(&sA[buf][0]), lx = lds_addr(&sX[buf][0]);
        async_cp16(la + r0 * 80 + s0, gA, r0 * kb + s0);
        async_cp16(la + r1 * 80 + s1, gA, r1 * kb + s1);
        async_cp16(lx + r0 * 80 + s0, gX, r0 * kb + s0);
        async_cp16(lx + r1 * 80 + s1, gX, r1 * kb + s1);
    };

    v8f acc[4][2] = {};
    issue(0, 0);
    int cur = 0;
    for (int k0 = 0; k0 < K; k0 += 32) {
        wait_async0();
        __syncthreads();
        if (k0 + 32 < K) issue(cur ^ 1, k0 + 32);

        v16h afr[4], bfr[2];
        #pragma unroll
        for (int fm = 0; fm < 4; ++fm) {
            const h16* p = &sA[cur][(wm * 64 + fm * 16 + rrow) * 40 + rsel];
            __builtin_memcpy(&afr[fm], p, 16);
            __builtin_memcpy((char*)&afr[fm] + 16, p + 16, 16);
        }
        #pragma unroll
        for (int fn = 0; fn < 2; ++fn) {
            const h16* p = &sX[cur][(wn * 32 + fn * 16 + rrow) * 40 + rsel];
            __builtin_memcpy(&bfr[fn], p, 16);
            __builtin_memcpy((char*)&bfr[fn] + 16, p + 16, 16);
        }
        #pragma unroll
        for (int fm = 0; fm < 4; ++fm)
            #pragma unroll
            for (int fn = 0; fn < 2; ++fn)
                acc[fm][fn] = __builtin_amdgcn_wmma_f32_16x16x32_f16(
                    false, afr[fm], false, bfr[fn], (short)0, acc[fm][fn], false, false);
        cur ^= 1;
    }

    #pragma unroll
    for (int fm = 0; fm < 4; ++fm) {
        #pragma unroll
        for (int fn = 0; fn < 2; ++fn) {
            const int mb = m0 + wm * 64 + fm * 16 + rsel;
            const int nn = n0 + wn * 32 + fn * 16 + rrow;
            if (TRANSOUT) {
                v8h o;
                #pragma unroll
                for (int v = 0; v < 8; ++v) {
                    float val = acc[fm][fn][v];
                    o[v] = (EPI == 1) ? ((val >= TH_ATTN) ? (h16)1.0f : (h16)0.0f)
                                      : (h16)val;
                }
                *(v8h*)&Y[((size_t)b * Ncols + nn) * M + mb] = o;
            } else {
                #pragma unroll
                for (int v = 0; v < 8; ++v) {
                    float val = acc[fm][fn][v];
                    Y[((size_t)b * M + mb + v) * Ncols + nn] =
                        (EPI == 1) ? ((val >= TH_ATTN) ? (h16)1.0f : (h16)0.0f)
                                   : (h16)val;
                }
            }
        }
    }
}

__global__ __launch_bounds__(256) void kv_wmma(const unsigned char* __restrict__ Ks,
                                               const unsigned char* __restrict__ Vs,
                                               h16* __restrict__ KV)
{
    __shared__ unsigned char sK[2][128 * 80];
    __shared__ unsigned char sV[2][128 * 80];
    const int b    = blockIdx.z;
    const int m0   = blockIdx.y * 128;
    const int n0   = blockIdx.x * 128;
    const int t    = threadIdx.x;
    const int lane = t & 31;
    const int wave = t >> 5;
    const int wm   = wave >> 2;
    const int wn   = wave & 3;
    const int rrow = lane & 15;
    const int rsel = (lane < 16) ? 0 : 8;

    const unsigned r0 = (unsigned)t >> 2,         s0 = ((unsigned)t & 3) * 16;
    const unsigned r1 = (unsigned)(t + 256) >> 2, s1 = ((unsigned)(t + 256) & 3) * 16;

    auto issue = [&](int buf, int k0) {
        const char* gK = (const char*)&Ks[((size_t)b * C_ + m0) * N_ + k0];
        const char* gV = (const char*)&Vs[((size_t)b * C_ + n0) * N_ + k0];
        unsigned lk = lds_addr(&sK[buf][0]), lv = lds_addr(&sV[buf][0]);
        async_cp16(lk + r0 * 80 + s0, gK, r0 * (unsigned)N_ + s0);
        async_cp16(lk + r1 * 80 + s1, gK, r1 * (unsigned)N_ + s1);
        async_cp16(lv + r0 * 80 + s0, gV, r0 * (unsigned)N_ + s0);
        async_cp16(lv + r1 * 80 + s1, gV, r1 * (unsigned)N_ + s1);
    };

    v8i acc[4][2] = {};
    issue(0, 0);
    int cur = 0;
    for (int k0 = 0; k0 < N_; k0 += 64) {
        wait_async0();
        __syncthreads();
        if (k0 + 64 < N_) issue(cur ^ 1, k0 + 64);

        v8i afr[4], bfr[2];
        #pragma unroll
        for (int fm = 0; fm < 4; ++fm) {
            const unsigned char* p = &sK[cur][(wm * 64 + fm * 16 + rrow) * 80 + rsel];
            __builtin_memcpy(&afr[fm], p, 8);
            __builtin_memcpy((char*)&afr[fm] + 8,  p + 16, 8);
            __builtin_memcpy((char*)&afr[fm] + 16, p + 32, 8);
            __builtin_memcpy((char*)&afr[fm] + 24, p + 48, 8);
        }
        #pragma unroll
        for (int fn = 0; fn < 2; ++fn) {
            const unsigned char* p = &sV[cur][(wn * 32 + fn * 16 + rrow) * 80 + rsel];
            __builtin_memcpy(&bfr[fn], p, 8);
            __builtin_memcpy((char*)&bfr[fn] + 8,  p + 16, 8);
            __builtin_memcpy((char*)&bfr[fn] + 16, p + 32, 8);
            __builtin_memcpy((char*)&bfr[fn] + 24, p + 48, 8);
        }
        #pragma unroll
        for (int fm = 0; fm < 4; ++fm)
            #pragma unroll
            for (int fn = 0; fn < 2; ++fn)
                acc[fm][fn] = __builtin_amdgcn_wmma_i32_16x16x64_iu8(
                    false, afr[fm], false, bfr[fn], acc[fm][fn], false, false);
        cur ^= 1;
    }

    #pragma unroll
    for (int fm = 0; fm < 4; ++fm) {
        #pragma unroll
        for (int fn = 0; fn < 2; ++fn) {
            const int cb = m0 + wm * 64 + fm * 16 + rsel;
            const int d  = n0 + wn * 32 + fn * 16 + rrow;
            v8h o;
            #pragma unroll
            for (int v = 0; v < 8; ++v)
                o[v] = (h16)(float)acc[fm][fn][v];
            *(v8h*)&KV[((size_t)b * C_ + d) * C_ + cb] = o;
        }
    }
}

__global__ void cvt_h16(const float* __restrict__ s, h16* __restrict__ d, int n)
{
    int i = blockIdx.x * 256 + threadIdx.x;
    if (i < n) d[i] = (h16)s[i];
}

__global__ __launch_bounds__(256) void transpose_x(const float* __restrict__ x,
                                                   h16* __restrict__ xT)
{
    __shared__ h16 tile[32][33];
    const int b  = blockIdx.z;
    const int n0 = blockIdx.x * 32, c0 = blockIdx.y * 32;
    const int tx = threadIdx.x & 31, ty = threadIdx.x >> 5;
    #pragma unroll
    for (int j = 0; j < 4; ++j) {
        int c = ty + j * 8;
        tile[c][tx] = (h16)x[((size_t)b * C_ + c0 + c) * N_ + n0 + tx];
    }
    __syncthreads();
    #pragma unroll
    for (int j = 0; j < 4; ++j) {
        int n = ty + j * 8;
        xT[((size_t)b * N_ + n0 + n) * C_ + c0 + tx] = tile[tx][n];
    }
}

template<bool TLAY>
__global__ __launch_bounds__(256) void bnstats(const h16* __restrict__ Y,
                                               const float* __restrict__ gamma,
                                               const float* __restrict__ beta,
                                               float* __restrict__ scale,
                                               float* __restrict__ shift)
{
    const int ch = blockIdx.x;
    const int t  = threadIdx.x;
    float s = 0.f, s2 = 0.f;
    for (int i = t; i < B_ * N_; i += 256) {
        size_t a = TLAY ? ((size_t)i * C_ + ch)
                        : (((size_t)(i >> 10) * C_ + ch) * N_ + (i & (N_ - 1)));
        float v = (float)Y[a];
        s += v; s2 += v * v;
    }
    __shared__ float rs[256], rq[256];
    rs[t] = s; rq[t] = s2;
    __syncthreads();
    for (int off = 128; off; off >>= 1) {
        if (t < off) { rs[t] += rs[t + off]; rq[t] += rq[t + off]; }
        __syncthreads();
    }
    if (t == 0) {
        const float inv = 1.0f / (float)(B_ * N_);
        float mean = rs[0] * inv;
        float var  = rq[0] * inv - mean * mean;
        float sc   = gamma[ch] * rsqrtf(var + 1e-5f);
        scale[ch] = sc;
        shift[ch] = beta[ch] - mean * sc;
    }
}

__global__ void spike_h16_T(const h16* __restrict__ Y, const float* __restrict__ sc,
                            const float* __restrict__ sh, h16* __restrict__ S)
{
    size_t i = (size_t)blockIdx.x * 256 + threadIdx.x;
    int ch = (int)(i & (C_ - 1));
    float v = sc[ch] * (float)Y[i] + sh[ch];
    S[i] = (v >= TH_BN) ? (h16)1.0f : (h16)0.0f;
}
__global__ void spike_u8(const h16* __restrict__ Y, const float* __restrict__ sc,
                         const float* __restrict__ sh, unsigned char* __restrict__ S)
{
    size_t i = (size_t)blockIdx.x * 256 + threadIdx.x;
    int ch = (int)((i >> 10) & (C_ - 1));
    float v = sc[ch] * (float)Y[i] + sh[ch];
    S[i] = (v >= TH_BN) ? 1 : 0;
}
__global__ void final_f32(const h16* __restrict__ YT, const float* __restrict__ sc,
                          const float* __restrict__ sh, float* __restrict__ out)
{
    size_t i = (size_t)blockIdx.x * 256 + threadIdx.x;
    int n  = (int)(i & (N_ - 1));
    int ch = (int)((i >> 10) & (C_ - 1));
    int b  = (int)(i >> 19);
    float v = sc[ch] * (float)YT[((size_t)b * N_ + n) * C_ + ch] + sh[ch];
    out[i] = (v >= TH_BN) ? 1.0f : 0.0f;
}

extern "C" void kernel_launch(void* const* d_in, const int* in_sizes, int n_in,
                              void* d_out, int out_size, void* d_ws, size_t ws_size,
                              hipStream_t stream)
{
    (void)in_sizes; (void)n_in; (void)out_size; (void)ws_size;
    const float* x  = (const float*)d_in[0];
    const float* qw = (const float*)d_in[1];
    const float* qg = (const float*)d_in[2];
    const float* qb = (const float*)d_in[3];
    const float* kw = (const float*)d_in[4];
    const float* kg = (const float*)d_in[5];
    const float* kb = (const float*)d_in[6];
    const float* vw = (const float*)d_in[7];
    const float* vg = (const float*)d_in[8];
    const float* vb = (const float*)d_in[9];
    const float* pw = (const float*)d_in[10];
    const float* pg = (const float*)d_in[11];
    const float* pb = (const float*)d_in[12];

    char* ws = (char*)d_ws;
    const size_t MB = 1ull << 20;
    h16* xT  = (h16*)(ws + 0 * MB);
    h16* yqT = (h16*)(ws + 32 * MB);
    h16* yk  = (h16*)(ws + 64 * MB);
    h16* yv  = (h16*)(ws + 96 * MB);
    h16* qsT = (h16*)(ws + 128 * MB);
    unsigned char* ksp = (unsigned char*)(ws + 160 * MB);
    unsigned char* vsp = (unsigned char*)(ws + 176 * MB);
    h16* kvh   = (h16*)(ws + 32 * MB);
    h16* attnT = (h16*)(ws + 64 * MB);
    h16* outT  = (h16*)(ws + 96 * MB);
    h16* whq = (h16*)(ws + 192 * MB);
    h16* whk = whq + (size_t)C_ * C_;
    h16* whv = whk + (size_t)C_ * C_;
    h16* whp = whv + (size_t)C_ * C_;
    float* st = (float*)(ws + 194 * MB);
    float *sc_q = st + 0,    *sh_q = st + 512;
    float *sc_k = st + 1024, *sh_k = st + 1536;
    float *sc_v = st + 2048, *sh_v = st + 2560;
    float *sc_p = st + 3072, *sh_p = st + 3584;

    dim3 blk(256);
    dim3 gg(N_ / 128, C_ / 128, B_);
    dim3 gkv(C_ / 128, C_ / 128, B_);
    dim3 gt(N_ / 32, C_ / 32, B_);
    dim3 ge((B_ * C_ * N_) / 256);

    const int WN = C_ * C_;
    cvt_h16<<<dim3((WN + 255) / 256), blk, 0, stream>>>(qw, whq, WN);
    cvt_h16<<<dim3((WN + 255) / 256), blk, 0, stream>>>(kw, whk, WN);
    cvt_h16<<<dim3((WN + 255) / 256), blk, 0, stream>>>(vw, whv, WN);
    cvt_h16<<<dim3((WN + 255) / 256), blk, 0, stream>>>(pw, whp, WN);
    transpose_x<<<gt, blk, 0, stream>>>(x, xT);

    gemm_wmma<false, 0, true ><<<gg, blk, 0, stream>>>(whq, xT, yqT, C_, C_, N_);
    gemm_wmma<false, 0, false><<<gg, blk, 0, stream>>>(whk, xT, yk,  C_, C_, N_);
    gemm_wmma<false, 0, false><<<gg, blk, 0, stream>>>(whv, xT, yv,  C_, C_, N_);

    bnstats<true ><<<dim3(C_), blk, 0, stream>>>(yqT, qg, qb, sc_q, sh_q);
    bnstats<false><<<dim3(C_), blk, 0, stream>>>(yk,  kg, kb, sc_k, sh_k);
    bnstats<false><<<dim3(C_), blk, 0, stream>>>(yv,  vg, vb, sc_v, sh_v);

    spike_h16_T<<<ge, blk, 0, stream>>>(yqT, sc_q, sh_q, qsT);
    spike_u8   <<<ge, blk, 0, stream>>>(yk,  sc_k, sh_k, ksp);
    spike_u8   <<<ge, blk, 0, stream>>>(yv,  sc_v, sh_v, vsp);

    kv_wmma<<<gkv, blk, 0, stream>>>(ksp, vsp, kvh);

    gemm_wmma<true,  1, true><<<gg, blk, 0, stream>>>(kvh, qsT, attnT, C_, C_, N_);
    gemm_wmma<false, 0, true><<<gg, blk, 0, stream>>>(whp, attnT, outT, C_, C_, N_);

    bnstats<true><<<dim3(C_), blk, 0, stream>>>(outT, pg, pb, sc_p, sh_p);
    final_f32<<<ge, blk, 0, stream>>>(outT, sc_p, sh_p, (float*)d_out);
}
